// TransformerEncoder_71863392797374
// MI455X (gfx1250) — matrix operand order repaired and checked by reading, not compiled, not run on hardware
//
#include <hip/hip_runtime.h>
#include <cstdint>
#include <cstddef>

typedef unsigned short u16;
typedef __attribute__((ext_vector_type(16))) __bf16 v16bf;
typedef __attribute__((ext_vector_type(8)))  float  v8f;

constexpr int kB = 32, kL = 448, kD = 256, kH = 8, kDK = 32, kFFD = 1024, kNL = 12;
constexpr int kM = kB * kL;                  // 14336 rows
constexpr float kScale = 0.17677669529663687f; // 1/sqrt(32)

union Frag { v16bf v; uint4 q[2]; };

__device__ __forceinline__ u16 f2bf(float f) {
  unsigned int u = __float_as_uint(f);
  u += 0x7FFFu + ((u >> 16) & 1u);           // round-to-nearest-even
  return (u16)(u >> 16);
}

__device__ __forceinline__ v8f vzero8() {
  v8f z;
#pragma unroll
  for (int i = 0; i < 8; ++i) z[i] = 0.0f;
  return z;
}

// --- CDNA5 async global->LDS (ASYNCcnt-tracked, no VGPR staging) --------------
__device__ __forceinline__ uint32_t lds_offset_of(const void* p) {
  // generic -> addrspace(3) explicit cast, then ptrtoint gives the LDS byte offset
  return (uint32_t)(size_t)(const __attribute__((address_space(3))) void*)p;
}

__device__ __forceinline__ void async_copy16(void* lds_dst, const void* gsrc) {
  uint32_t loff = lds_offset_of(lds_dst);
  unsigned long long ga = (unsigned long long)(size_t)gsrc;
  asm volatile("global_load_async_to_lds_b128 %0, %1, off"
               :: "v"(loff), "v"(ga) : "memory");
}

__device__ __forceinline__ void wait_async0() {
  asm volatile("s_wait_asynccnt 0x0" ::: "memory");
}

// ---------------------------------------------------------------------------
// Weight prep: fp32 W[K][N] (per layer) -> bf16 Wt[N][K]
// ---------------------------------------------------------------------------
__global__ __launch_bounds__(256)
void prep_w_kernel(const float* __restrict__ W, u16* __restrict__ Wt, int K, int N) {
  size_t per = (size_t)K * N;
  const float* Wl = W + per * blockIdx.y;
  u16* Wtl = Wt + per * blockIdx.y;
  size_t i = (size_t)blockIdx.x * 256 + threadIdx.x;
  if (i < per) {
    int k = (int)(i / N), n = (int)(i % N);
    Wtl[(size_t)n * K + k] = f2bf(Wl[i]);
  }
}

// ---------------------------------------------------------------------------
// x = x + pos_emb (broadcast over batch), writes running f32 buffer (d_out)
// ---------------------------------------------------------------------------
__global__ __launch_bounds__(256)
void add_pos_kernel(const float* __restrict__ x, const float* __restrict__ pos,
                    float* __restrict__ xw) {
  size_t i = (size_t)blockIdx.x * 256 + threadIdx.x;
  int ld = (int)(i % ((size_t)kL * kD));
  xw[i] = x[i] + pos[ld];
}

// ---------------------------------------------------------------------------
// LayerNorm over D=256, one wave per row, output bf16 (GEMM A operand)
// ---------------------------------------------------------------------------
__global__ __launch_bounds__(256)
void ln_kernel(const float* __restrict__ x, u16* __restrict__ out,
               const float* __restrict__ g, const float* __restrict__ b, int rows) {
  int wave = threadIdx.x >> 5, lane = threadIdx.x & 31;
  int row = blockIdx.x * 8 + wave;
  if (row >= rows) return;
  const float* xr = x + (size_t)row * kD;
  float v[8];
  float s = 0.f;
#pragma unroll
  for (int i = 0; i < 8; ++i) { v[i] = xr[lane + i * 32]; s += v[i]; }
#pragma unroll
  for (int off = 16; off > 0; off >>= 1) s += __shfl_xor(s, off, 32);
  float mu = s * (1.0f / kD);
  float var = 0.f;
#pragma unroll
  for (int i = 0; i < 8; ++i) { float d = v[i] - mu; var += d * d; }
#pragma unroll
  for (int off = 16; off > 0; off >>= 1) var += __shfl_xor(var, off, 32);
  var *= (1.0f / kD);
  float inv = rsqrtf(var + 1e-6f);
#pragma unroll
  for (int i = 0; i < 8; ++i) {
    int c = lane + i * 32;
    out[(size_t)row * kD + c] = f2bf((v[i] - mu) * inv * g[c] + b[c]);
  }
}

// ---------------------------------------------------------------------------
// Generic bf16 WMMA GEMM: out = A[M,K] @ Wt[N,K]^T + bias, fused epilogues.
// 128 threads (4 waves), 64x64x32 tile, each wave 32x32 (2x2 WMMA tiles).
// Double-buffered LDS fed by async global->LDS copies: one s_wait_asynccnt +
// one barrier per K-step; tile t+1 streams in while tile t is consumed.
// ---------------------------------------------------------------------------
enum { EP_QK = 0, EP_VT = 1, EP_RES = 2, EP_GELU = 3 };

__global__ __launch_bounds__(128)
void gemm_bf16_kernel(const u16* __restrict__ A, const u16* __restrict__ Bt,
                      const float* __restrict__ bias, void* __restrict__ dst,
                      int M, int N, int K, int mode) {
  constexpr int BM = 64, BN = 64, BK = 32, LDS = BK + 8; // padded stride (16B-aligned)
  __shared__ __align__(16) u16 sA[2][BM * LDS];
  __shared__ __align__(16) u16 sB[2][BN * LDS];

  const int tid  = threadIdx.x;
  const int lane = tid & 31;
  const int wave = tid >> 5;
  const int wm   = (wave >> 1) * 32;
  const int wn   = (wave & 1) * 32;
  const int half = lane >> 4;
  const int l16  = lane & 15;
  const int blockM = blockIdx.x * BM;
  const int blockN = blockIdx.y * BN;

  v8f acc[2][2];
#pragma unroll
  for (int i = 0; i < 2; ++i)
#pragma unroll
    for (int j = 0; j < 2; ++j) acc[i][j] = vzero8();

  // each thread owns two 16B chunks of the 64x32 tile (256 chunks / 128 threads)
  const int row0 = (tid + 0)   >> 2, seg0 = (tid + 0)   & 3;
  const int row1 = (tid + 128) >> 2, seg1 = (tid + 128) & 3;

  auto issue_tile = [&](int k0, int buf) {
    async_copy16(&sA[buf][row0 * LDS + seg0 * 8],
                 &A[(size_t)(blockM + row0) * K + k0 + seg0 * 8]);
    async_copy16(&sB[buf][row0 * LDS + seg0 * 8],
                 &Bt[(size_t)(blockN + row0) * K + k0 + seg0 * 8]);
    async_copy16(&sA[buf][row1 * LDS + seg1 * 8],
                 &A[(size_t)(blockM + row1) * K + k0 + seg1 * 8]);
    async_copy16(&sB[buf][row1 * LDS + seg1 * 8],
                 &Bt[(size_t)(blockN + row1) * K + k0 + seg1 * 8]);
  };

  issue_tile(0, 0);

  const int nIter = K / BK;
  for (int it = 0; it < nIter; ++it) {
    wait_async0();        // tile `it` fully landed in LDS (this wave's copies)
    __syncthreads();      // ...and everyone else's; prior reads of buf^1 done

    if (it + 1 < nIter) issue_tile((it + 1) * BK, (it + 1) & 1);

    const u16* bufA = sA[it & 1];
    const u16* bufB = sB[it & 1];
    Frag af[2], bf[2];
#pragma unroll
    for (int t = 0; t < 2; ++t) {
      // A fragment (16x32 bf16): lanes 0-15 K{0..7,16..23}, lanes 16-31 K{8..15,24..31}
      int ar = wm + t * 16 + l16;
      af[t].q[0] = *(const uint4*)(&bufA[ar * LDS + half * 8]);
      af[t].q[1] = *(const uint4*)(&bufA[ar * LDS + 16 + half * 8]);
      // B fragment (32x16): lane n holds column n, lanes 0-15 K{0..7,16..23}, lanes 16-31 K{8..15,24..31}
      int br = wn + t * 16 + l16;
      bf[t].q[0] = *(const uint4*)(&bufB[br * LDS + half * 8]);
      bf[t].q[1] = *(const uint4*)(&bufB[br * LDS + 16 + half * 8]);
    }
#pragma unroll
    for (int mt = 0; mt < 2; ++mt)
#pragma unroll
      for (int nt = 0; nt < 2; ++nt)
        acc[mt][nt] = __builtin_amdgcn_wmma_f32_16x16x32_bf16(
            false, af[mt].v, false, bf[nt].v, (short)0, acc[mt][nt], false, false);
  }

#pragma unroll
  for (int mt = 0; mt < 2; ++mt)
#pragma unroll
    for (int nt = 0; nt < 2; ++nt)
#pragma unroll
      for (int r = 0; r < 8; ++r) {
        int m = blockM + wm + mt * 16 + half * 8 + r;  // C layout: lanes>=16 hold M+8
        int n = blockN + wn + nt * 16 + l16;
        float val = acc[mt][nt][r] + bias[n];
        if (mode == EP_RES) {
          ((float*)dst)[(size_t)m * N + n] += val;
        } else if (mode == EP_GELU) {
          float ge = 0.5f * val * (1.0f + erff(val * 0.70710678118654752f));
          ((u16*)dst)[(size_t)m * N + n] = f2bf(ge);
        } else {
          int bb = m / kL, l = m - bb * kL;
          int h = n >> 5, dk = n & 31;
          size_t idx = (mode == EP_QK)
              ? ((((size_t)bb * kH + h) * kL + l) * kDK + dk)     // [B,H,L,DK]
              : ((((size_t)bb * kH + h) * kDK + dk) * kL + l);    // [B,H,DK,L]
          ((u16*)dst)[idx] = f2bf(val);
        }
      }
}

// ---------------------------------------------------------------------------
// Attention: one wave per (b,h,16-query rows). QK^T via WMMA (K=DK=32 exact),
// softmax in LDS, P (bf16) @ V via chained WMMA over 14 K-chunks.
// ---------------------------------------------------------------------------
__global__ __launch_bounds__(32)
void attn_kernel(const u16* __restrict__ q, const u16* __restrict__ k,
                 const u16* __restrict__ vT, const unsigned char* __restrict__ mask,
                 u16* __restrict__ o) {
  __shared__ __align__(16) float S[16][456];
  __shared__ __align__(16) u16   P[16][464];

  const int bh = blockIdx.x;           // b*H + h
  const int qt = blockIdx.y;           // query tile (0..27)
  const int b  = bh / kH, h = bh - b * kH;
  const int lane = threadIdx.x;
  const int half = lane >> 4, l16 = lane & 15;

  // Q fragment (A operand), rows qt*16..+15
  const u16* qp = q + ((size_t)bh * kL + qt * 16) * kDK;
  Frag af;
  af.q[0] = *(const uint4*)(qp + (size_t)l16 * kDK + half * 8);
  af.q[1] = *(const uint4*)(qp + (size_t)l16 * kDK + 16 + half * 8);

  for (int kt = 0; kt < kL / 16; ++kt) {
    // B fragment = K^T: lane n = key (kt*16+n), K-dim = dk contiguous in k[B,H,L,DK]
    const u16* kp = k + ((size_t)bh * kL + kt * 16) * kDK;
    Frag bfr;
    bfr.q[0] = *(const uint4*)(kp + (size_t)l16 * kDK + half * 8);
    bfr.q[1] = *(const uint4*)(kp + (size_t)l16 * kDK + 16 + half * 8);
    v8f c = vzero8();
    c = __builtin_amdgcn_wmma_f32_16x16x32_bf16(false, af.v, false, bfr.v,
                                                (short)0, c, false, false);
    int col = kt * 16 + l16;
    bool msk = mask[(size_t)b * kL + col] != 0;
#pragma unroll
    for (int r = 0; r < 8; ++r)
      S[half * 8 + r][col] = msk ? -10000.0f : c[r] * kScale;
  }
  __syncthreads();

  // Row softmax: 2 lanes per row, 224 columns each
  {
    int row = l16, seg = half;
    float mx = -3.4e38f;
    for (int c0 = 0; c0 < 224; ++c0) mx = fmaxf(mx, S[row][seg * 224 + c0]);
    mx = fmaxf(mx, __shfl_xor(mx, 16, 32));
    float sum = 0.f;
    for (int c0 = 0; c0 < 224; ++c0) {
      float e = __expf(S[row][seg * 224 + c0] - mx);
      S[row][seg * 224 + c0] = e;
      sum += e;
    }
    sum += __shfl_xor(sum, 16, 32);
    float inv = 1.0f / sum;
    for (int c0 = 0; c0 < 224; ++c0)
      P[row][seg * 224 + c0] = f2bf(S[row][seg * 224 + c0] * inv);
  }
  __syncthreads();

  // O = P[16,448] @ V[448,32] : N split into two 16-wide tiles, 14 K-chunks of 32
#pragma unroll
  for (int nt = 0; nt < 2; ++nt) {
    v8f c = vzero8();
    for (int kc = 0; kc < kL / 32; ++kc) {
      Frag a;
      a.q[0] = *(const uint4*)(&P[l16][kc * 32 + half * 8]);
      a.q[1] = *(const uint4*)(&P[l16][kc * 32 + 16 + half * 8]);
      const u16* vp = vT + ((size_t)bh * kDK + nt * 16 + l16) * kL + kc * 32 + half * 8;
      Frag bb;
      bb.q[0] = *(const uint4*)(vp);
      bb.q[1] = *(const uint4*)(vp + 16);
      c = __builtin_amdgcn_wmma_f32_16x16x32_bf16(false, a.v, false, bb.v,
                                                  (short)0, c, false, false);
    }
#pragma unroll
    for (int r = 0; r < 8; ++r) {
      size_t m = (size_t)b * kL + qt * 16 + half * 8 + r;
      int n = h * kDK + nt * 16 + l16;
      o[m * kD + n] = f2bf(c[r]);
    }
  }
}

// ---------------------------------------------------------------------------
// Host orchestration
// ---------------------------------------------------------------------------
extern "C" void kernel_launch(void* const* d_in, const int* in_sizes, int n_in,
                              void* d_out, int out_size, void* d_ws, size_t ws_size,
                              hipStream_t stream) {
  (void)in_sizes; (void)n_in; (void)out_size; (void)ws_size;

  const float* x    = (const float*)d_in[0];
  const unsigned char* mask = (const unsigned char*)d_in[1];
  const float* pos  = (const float*)d_in[2];
  const float* Wq   = (const float*)d_in[3];
  const float* bq   = (const float*)d_in[4];
  const float* Wk   = (const float*)d_in[5];
  const float* bk   = (const float*)d_in[6];
  const float* Wv   = (const float*)d_in[7];
  const float* bv   = (const float*)d_in[8];
  const float* Wo   = (const float*)d_in[9];
  const float* bo   = (const float*)d_in[10];
  const float* W1   = (const float*)d_in[11];
  const float* bf1  = (const float*)d_in[12];
  const float* W2   = (const float*)d_in[13];
  const float* bf2  = (const float*)d_in[14];
  const float* ln1g = (const float*)d_in[15];
  const float* ln1b = (const float*)d_in[16];
  const float* ln2g = (const float*)d_in[17];
  const float* ln2b = (const float*)d_in[18];

  float* xw = (float*)d_out;  // running f32 activation buffer == final output

  char* wsp = (char*)d_ws;
  auto alloc = [&](size_t bytes) -> char* {
    char* p = wsp; wsp += (bytes + 255) & ~(size_t)255; return p;
  };
  const size_t wDD  = (size_t)kNL * kD * kD;
  const size_t wDF  = (size_t)kNL * kD * kFFD;
  u16* WqT = (u16*)alloc(wDD * 2);
  u16* WkT = (u16*)alloc(wDD * 2);
  u16* WvT = (u16*)alloc(wDD * 2);
  u16* WoT = (u16*)alloc(wDD * 2);
  u16* W1T = (u16*)alloc(wDF * 2);
  u16* W2T = (u16*)alloc(wDF * 2);
  u16* nx  = (u16*)alloc((size_t)kM * kD * 2);
  u16* qb  = (u16*)alloc((size_t)kM * kD * 2);    // [B,H,L,DK]
  u16* kb  = (u16*)alloc((size_t)kM * kD * 2);    // [B,H,L,DK]
  u16* vT  = (u16*)alloc((size_t)kM * kD * 2);    // [B,H,DK,L]
  u16* ob  = (u16*)alloc((size_t)kM * kD * 2);    // [M,D]
  u16* hb  = (u16*)alloc((size_t)kM * kFFD * 2);  // [M,FFD]

  // --- weight prep (bf16 + transpose) ---
  {
    dim3 gdd((kD * kD + 255) / 256, kNL), gdf((kD * kFFD + 255) / 256, kNL);
    prep_w_kernel<<<gdd, 256, 0, stream>>>(Wq, WqT, kD, kD);
    prep_w_kernel<<<gdd, 256, 0, stream>>>(Wk, WkT, kD, kD);
    prep_w_kernel<<<gdd, 256, 0, stream>>>(Wv, WvT, kD, kD);
    prep_w_kernel<<<gdd, 256, 0, stream>>>(Wo, WoT, kD, kD);
    prep_w_kernel<<<gdf, 256, 0, stream>>>(W1, W1T, kD, kFFD);
    prep_w_kernel<<<gdf, 256, 0, stream>>>(W2, W2T, kFFD, kD);
  }

  // --- x = x + pos ---
  add_pos_kernel<<<(kM * kD) / 256, 256, 0, stream>>>(x, pos, xw);

  const dim3 gemmBlk(128);
  const dim3 gDD(kM / 64, kD / 64);      // (224, 4)
  const dim3 gDF(kM / 64, kFFD / 64);    // (224, 16)
  const dim3 gAttn(kB * kH, kL / 16);    // (256, 28)

  for (int i = 0; i < kNL; ++i) {
    const size_t oDD = (size_t)i * kD * kD;
    const size_t oDF = (size_t)i * kD * kFFD;

    // LN1 -> nx (bf16)
    ln_kernel<<<kM / 8, 256, 0, stream>>>(xw, nx, ln1g + i * kD, ln1b + i * kD, kM);

    // Q, K, V projections
    gemm_bf16_kernel<<<gDD, gemmBlk, 0, stream>>>(nx, WqT + oDD, bq + i * kD, qb,
                                                  kM, kD, kD, EP_QK);
    gemm_bf16_kernel<<<gDD, gemmBlk, 0, stream>>>(nx, WkT + oDD, bk + i * kD, kb,
                                                  kM, kD, kD, EP_QK);
    gemm_bf16_kernel<<<gDD, gemmBlk, 0, stream>>>(nx, WvT + oDD, bv + i * kD, vT,
                                                  kM, kD, kD, EP_VT);

    // Attention -> ob (bf16 [M,D])
    attn_kernel<<<gAttn, 32, 0, stream>>>(qb, kb, vT, mask, ob);

    // x += ob @ Wo + bo
    gemm_bf16_kernel<<<gDD, gemmBlk, 0, stream>>>(ob, WoT + oDD, bo + i * kD, xw,
                                                  kM, kD, kD, EP_RES);

    // LN2 -> nx
    ln_kernel<<<kM / 8, 256, 0, stream>>>(xw, nx, ln2g + i * kD, ln2b + i * kD, kM);

    // h = gelu(nx @ W1 + bf1)
    gemm_bf16_kernel<<<gDF, gemmBlk, 0, stream>>>(nx, W1T + oDF, bf1 + i * kFFD, hb,
                                                  kM, kFFD, kD, EP_GELU);

    // x += h @ W2 + bf2
    gemm_bf16_kernel<<<gDD, gemmBlk, 0, stream>>>(hb, W2T + oDF, bf2 + i * kD, xw,
                                                  kM, kD, kFFD, EP_RES);
  }
}
